// AutoregressivePrior_66949950210695
// MI455X (gfx1250) — matrix operand order repaired and checked by reading, not compiled, not run on hardware
//
#include <hip/hip_runtime.h>

typedef __attribute__((ext_vector_type(16))) _Float16 v16h;
typedef __attribute__((ext_vector_type(8)))  float    v8f;
typedef __attribute__((ext_vector_type(4)))  float    v4f;

#define HID    256
#define KSTEPS 7
#define ZM     64
#define BATCH  32768
#define BCHUNK 128   // batch chunks per (tensor,k) pair in phase 2

__device__ __forceinline__ float sigmoidf_(float x) {
    return 1.0f / (1.0f + __expf(-x));
}

// ---------------------------------------------------------------------------
// Phase 1: one workgroup (8 wave32) runs the sequential LSTM + projections.
// All matmul work is done with v_wmma_f32_16x16x32_f16.
//   gates = [x;h] (K=512) x [W_ih | W_hh]^T   -> 64 N-tiles of 16, 8 per wave
//   loc/scale = p_z(16x256, rows 7..15 = 0) x W^T -> 4 N-tiles each
// A-fragment K order is contiguous (16-bit A, wave32):
//   lanes 0-15 hold K = {k+0..15}, lanes 16-31 hold {k+16..31}
// B-fragment uses the same contiguous K order as A: VGPR v packs rows K=2v,2v+1
//   (lanes 0-15) / K=2v+16,2v+17 (lanes 16-31), column = lane%16.
// ---------------------------------------------------------------------------
__global__ __launch_bounds__(256) void lstm_prior_phase1(
    const float* __restrict__ zm1,
    const float* __restrict__ W_ih, const float* __restrict__ W_hh,
    const float* __restrict__ b_ih, const float* __restrict__ b_hh,
    const float* __restrict__ W_loc, const float* __restrict__ b_loc,
    const float* __restrict__ W_scale, const float* __restrict__ b_scale,
    float* __restrict__ ws)
{
    __shared__ float    pz[16 * HID];      // p_z padded to 16 rows (f32)
    __shared__ _Float16 pz16[16 * HID];    // f16 copy for projection WMMA
    __shared__ _Float16 xh16[2 * HID];     // [x ; h] in f16, K=512
    __shared__ float    gates[4 * HID];
    __shared__ float    hbuf[HID];
    __shared__ float    cbuf[HID];

    const int  tid    = threadIdx.x;
    const int  wave   = tid >> 5;
    const int  lane   = tid & 31;
    const int  lane16 = lane & 15;
    const bool hiHalf = (lane >= 16);
    const int  koff   = hiHalf ? 8 : 0;    // unused (A-frag uses kg)
    const int  kg     = hiHalf ? 16 : 0;   // B-frag K sub-offset

    for (int i = tid; i < 16 * HID; i += 256) pz[i] = 0.0f;
    __syncthreads();
    pz[tid]  = zm1[tid];                   // row 0 = zm_1
    hbuf[tid] = 0.0f;
    cbuf[tid] = 0.0f;
    __syncthreads();

    // ---------------- sequential LSTM, 6 steps --------------------------
    for (int step = 0; step < KSTEPS - 1; ++step) {
        xh16[tid]       = (_Float16)pz[step * HID + tid];
        xh16[HID + tid] = (_Float16)hbuf[tid];
        __syncthreads();

        for (int t = 0; t < 8; ++t) {                 // 8 N-tiles per wave
            const int nbase = (wave * 8 + t) * 16;
            const int n     = nbase + lane16;
            v8f acc = {};
            #pragma unroll
            for (int kc = 0; kc < 16; ++kc) {         // K = 512 in chunks of 32
                const int kbase = kc * 32;
                v16h a;
                #pragma unroll
                for (int j = 0; j < 8; ++j) {
                    a[j]     = xh16[kbase + kg + j];
                    a[8 + j] = xh16[kbase + kg + 8 + j];
                }
                const float* Wrow = (kbase < HID)
                                  ? (W_ih + n * HID + kbase)
                                  : (W_hh + n * HID + (kbase - HID));
                v16h b;
                #pragma unroll
                for (int v = 0; v < 8; ++v) {
                    b[2 * v]     = (_Float16)Wrow[2 * v + kg];
                    b[2 * v + 1] = (_Float16)Wrow[2 * v + 1 + kg];
                }
                acc = __builtin_amdgcn_wmma_f32_16x16x32_f16(
                          false, a, false, b, (short)0, acc, false, false);
            }
            // A rows replicated -> every D row equals the gate row; take M=0
            if (!hiHalf) gates[n] = acc[0] + b_ih[n] + b_hh[n];
        }
        __syncthreads();

        const float gi = gates[tid];
        const float gf = gates[HID + tid];
        const float gg = gates[2 * HID + tid];
        const float go = gates[3 * HID + tid];
        const float c  = sigmoidf_(gf) * cbuf[tid] + sigmoidf_(gi) * tanhf(gg);
        const float h  = sigmoidf_(go) * tanhf(c);
        cbuf[tid] = c;
        hbuf[tid] = h;
        pz[(step + 1) * HID + tid] = h;   // z == h_new, also next input
        __syncthreads();
    }

    for (int i = tid; i < 16 * HID; i += 256) pz16[i] = (_Float16)pz[i];
    __syncthreads();

    // ---------------- projections: loc & scale, [16x256]x[256x64] -------
    const int    which = wave >> 2;                 // 0 = loc, 1 = scale
    const int    nbase = (wave & 3) * 16;
    const float* Wp    = which ? W_scale : W_loc;   // [64,256] row-major
    const float* bp    = which ? b_scale : b_loc;
    const int    n     = nbase + lane16;

    v8f acc = {};
    #pragma unroll
    for (int kc = 0; kc < 8; ++kc) {                // K = 256
        const int kbase = kc * 32;
        v16h a;
        #pragma unroll
        for (int j = 0; j < 8; ++j) {
            a[j]     = pz16[lane16 * HID + kbase + kg + j];
            a[8 + j] = pz16[lane16 * HID + kbase + kg + 8 + j];
        }
        const float* Wrow = Wp + n * HID + kbase;
        v16h b;
        #pragma unroll
        for (int v = 0; v < 8; ++v) {
            b[2 * v]     = (_Float16)Wrow[2 * v + kg];
            b[2 * v + 1] = (_Float16)Wrow[2 * v + 1 + kg];
        }
        acc = __builtin_amdgcn_wmma_f32_16x16x32_f16(
                  false, a, false, b, (short)0, acc, false, false);
    }
    // D rows M=0..6 live in acc[0..6] for lanes 0-15 (N = lane)
    if (!hiHalf) {
        float* dst = ws + which * (KSTEPS * ZM);
        #pragma unroll
        for (int r = 0; r < KSTEPS; ++r)
            dst[r * ZM + n] = acc[r] + bp[n];
    }
}

// ---------------------------------------------------------------------------
// Phase 2: broadcast 2 x [7,64] over batch -> 117 MB of nontemporal stores.
// One block = (tensor, k, batch-chunk of 256 rows); each thread owns one
// float4 of the row (read once from L2-resident ws) and streams 16 stores.
// ---------------------------------------------------------------------------
__global__ __launch_bounds__(256) void broadcast_phase2(
    const float* __restrict__ ws, float* __restrict__ out)
{
    const int tid   = threadIdx.x;
    const int bx    = blockIdx.x;
    const int which = bx / (KSTEPS * BCHUNK);
    const int rem   = bx % (KSTEPS * BCHUNK);
    const int k     = rem / BCHUNK;
    const int chunk = rem % BCHUNK;

    const int rows_per_chunk = BATCH / BCHUNK;   // 256
    const int jvec = tid & 15;                   // which float4 of the 64-f row
    const int brow = tid >> 4;                   // 0..15

    const v4f val =
        ((const v4f*)(ws + which * (KSTEPS * ZM) + k * ZM))[jvec];

    v4f* outp = (v4f*)out + (size_t)which * ((size_t)KSTEPS * BATCH * (ZM / 4));
    const size_t row0 =
        (size_t)k * BATCH + (size_t)chunk * rows_per_chunk + brow;

    #pragma unroll
    for (int i = 0; i < rows_per_chunk / 16; ++i) {
        __builtin_nontemporal_store(
            val, &outp[(row0 + (size_t)i * 16) * (ZM / 4) + jvec]);
    }
}

extern "C" void kernel_launch(void* const* d_in, const int* in_sizes, int n_in,
                              void* d_out, int out_size, void* d_ws, size_t ws_size,
                              hipStream_t stream)
{
    const float* zm1     = (const float*)d_in[0];
    const float* W_ih    = (const float*)d_in[1];
    const float* W_hh    = (const float*)d_in[2];
    const float* b_ih    = (const float*)d_in[3];
    const float* b_hh    = (const float*)d_in[4];
    const float* W_loc   = (const float*)d_in[5];
    const float* b_loc   = (const float*)d_in[6];
    const float* W_scale = (const float*)d_in[7];
    const float* b_scale = (const float*)d_in[8];
    // d_in[9] = K (7), d_in[10] = batch_size (32768): fixed by the reference.

    float* ws = (float*)d_ws;   // [2][7][64] f32 = 3584 B

    lstm_prior_phase1<<<1, 256, 0, stream>>>(
        zm1, W_ih, W_hh, b_ih, b_hh, W_loc, b_loc, W_scale, b_scale, ws);

    broadcast_phase2<<<2 * KSTEPS * BCHUNK, 256, 0, stream>>>(
        ws, (float*)d_out);
}
